// CustomLinearRNN_29867202577007
// MI455X (gfx1250) — compile-verified, matrix operand order checked by reading, not run on hardware
//
#include <hip/hip_runtime.h>

// Linear RNN scan on gfx1250 (MI455X), split into:
//  Kernel 1 (serial, latency-critical): h_t = A3 @ x_t + A4 @ h_{t-1}
//    8 workgroups (one per 16-col batch tile) x 8 waves (one per 16-row h tile).
//    One barrier per step; h double-buffered in LDS (bf16); x staged one step
//    ahead into a double-buffered LDS tile; WMMA dep chain split in two.
//  Kernel 2 (parallel, bandwidth-bound): known_t = A1 @ x_t + A2 @ h_{t-1}
//    grid over all timesteps; reads hidden_seq produced by kernel 1.

typedef __attribute__((ext_vector_type(16))) __bf16 bf16x16;
typedef __attribute__((ext_vector_type(8)))  float  f32x8;

#define T_STEPS 4096
#define KDIM 64
#define HDIM 128
#define BDIM 128
#define BT 16            // batch-tile columns (WMMA N)
#define TCHUNK 8         // timesteps per block in the known kernel

union FragCast { uint4 u[2]; bf16x16 v; };

// B-fragment (32x16 bf16, K x N) from LDS laid out [n][k] (row stride `stride`).
// Lane l: n = l&15, half = l>>4; per-lane 16 elems = runs of 8 at kbase+half*8
// and kbase+16+half*8 (ISA 7.12.2, 16-bit operand layout).
__device__ __forceinline__ bf16x16 lds_bfrag(const __bf16* s, int n, int stride,
                                             int kbase, int half) {
  FragCast f;
  const __bf16* p = s + n * stride + kbase + half * 8;
  f.u[0] = *(const uint4*)(p);
  f.u[1] = *(const uint4*)(p + 16);
  return f.v;
}

// A-fragment (16x32 bf16, M x K) from global fp32 matrix; M = lane&15.
__device__ __forceinline__ bf16x16 gld_afrag(const float* __restrict__ A, int row,
                                             int stride, int kbase, int half) {
  bf16x16 f;
  const float* p0 = A + (size_t)row * stride + kbase + half * 8;
#pragma unroll
  for (int j = 0; j < 8; ++j) {
    f[j]     = (__bf16)p0[j];
    f[j + 8] = (__bf16)p0[j + 16];
  }
  return f;
}

// Stage one x_t tile [KDIM rows][BT cols] fp32 -> bf16 LDS [n][k].
// Exactly one float4 per thread with 256 threads.
__device__ __forceinline__ void stage_x_tile(const float* __restrict__ xt,
                                             __bf16* dst, int i) {
  int kk = i >> 2;            // row k: 4 float4s per 16-wide row
  int n4 = (i & 3) * 4;       // starting column
  float4 v = *(const float4*)(xt + (size_t)kk * BDIM + n4);
  dst[(n4 + 0) * KDIM + kk] = (__bf16)v.x;
  dst[(n4 + 1) * KDIM + kk] = (__bf16)v.y;
  dst[(n4 + 2) * KDIM + kk] = (__bf16)v.z;
  dst[(n4 + 3) * KDIM + kk] = (__bf16)v.w;
}

// ---------------------------------------------------------------------------
// Kernel 1: sequential hidden scan. 8 blocks x 256 threads (8 waves).
// ---------------------------------------------------------------------------
__global__ __launch_bounds__(256, 1)
void rnn_hidden_scan(const float* __restrict__ x,   // [T,K,B]
                     const float* __restrict__ h0,  // [H,B]
                     const float* __restrict__ A3,  // [H,K]
                     const float* __restrict__ A4,  // [H,H]
                     float* __restrict__ hidden_out)  // [T,H,B]
{
  __shared__ __align__(16) __bf16 sh_h[2][BT * HDIM];  // [n][k] double buffered
  __shared__ __align__(16) __bf16 sh_x[2][BT * KDIM];  // staged one step ahead

  const int tid  = threadIdx.x;
  const int wave = tid >> 5;
  const int lane = tid & 31;
  const int half = lane >> 4;
  const int n    = lane & 15;
  const int b0   = blockIdx.x * BT;
  const int m0   = wave * 16;

  // Per-wave A3/A4 row-tile fragments, resident for all 4096 steps.
  bf16x16 aX[2], aH[4];
#pragma unroll
  for (int c = 0; c < 2; ++c) aX[c] = gld_afrag(A3, m0 + n, KDIM, c * 32, half);
#pragma unroll
  for (int c = 0; c < 4; ++c) aH[c] = gld_afrag(A4, m0 + n, HDIM, c * 32, half);

  // h_{-1} into buffer 0; x_0 into x buffer 0.
  for (int i = tid; i < BT * HDIM; i += 256) {
    int nn = i / HDIM, kk = i % HDIM;
    sh_h[0][nn * HDIM + kk] = (__bf16)h0[(size_t)kk * BDIM + b0 + nn];
  }
  stage_x_tile(x + b0, sh_x[0], tid);
  __syncthreads();

  for (int t = 0; t < T_STEPS; ++t) {
    const int cur = t & 1, nxt = cur ^ 1;
    const __bf16* hc = sh_h[cur];
    __bf16*       hn = sh_h[nxt];

    // x-part: independent of h, can start immediately.
    bf16x16 bx0 = lds_bfrag(sh_x[cur], n, KDIM, 0, half);
    bf16x16 bx1 = lds_bfrag(sh_x[cur], n, KDIM, 32, half);
    // Preload all h fragments so the ds_load_b128s are all in flight at once.
    bf16x16 bh[4];
#pragma unroll
    for (int c = 0; c < 4; ++c) bh[c] = lds_bfrag(hc, n, HDIM, c * 32, half);

    // Two accumulator chains: h-dependency depth is 2 WMMAs, not 4.
    f32x8 accA = {}, accB = {};
    accA = __builtin_amdgcn_wmma_f32_16x16x32_bf16(false, aX[0], false, bx0,
                                                   (short)0, accA, false, false);
    accA = __builtin_amdgcn_wmma_f32_16x16x32_bf16(false, aX[1], false, bx1,
                                                   (short)0, accA, false, false);
    accB = __builtin_amdgcn_wmma_f32_16x16x32_bf16(false, aH[0], false, bh[0],
                                                   (short)0, accB, false, false);
    accA = __builtin_amdgcn_wmma_f32_16x16x32_bf16(false, aH[1], false, bh[1],
                                                   (short)0, accA, false, false);
    accB = __builtin_amdgcn_wmma_f32_16x16x32_bf16(false, aH[2], false, bh[2],
                                                   (short)0, accB, false, false);
    accA = __builtin_amdgcn_wmma_f32_16x16x32_bf16(false, aH[3], false, bh[3],
                                                   (short)0, accA, false, false);

    // While WMMAs are in flight: stage x_{t+1} into the other x buffer and
    // prefetch further ahead (global_prefetch_b8).
    if (t + 1 < T_STEPS)
      stage_x_tile(x + (size_t)(t + 1) * KDIM * BDIM + b0, sh_x[nxt], tid);
    if (t + 8 < T_STEPS && tid < KDIM)
      __builtin_prefetch(x + (size_t)(t + 8) * KDIM * BDIM + (size_t)tid * BDIM + b0, 0, 1);

    f32x8 acc = accA + accB;

    // fp32 result tile -> global. C/D layout: VGPR v -> m = half*8 + v.
    float* outp = hidden_out + (size_t)t * HDIM * BDIM + (size_t)m0 * BDIM + b0;
#pragma unroll
    for (int v = 0; v < 8; ++v)
      outp[(size_t)(half * 8 + v) * BDIM + n] = acc[v];

    // Publish bf16 h_t into next LDS buffer: one 16B ds_store per lane.
    union { uint4 u; __bf16 h[8]; } st;
#pragma unroll
    for (int v = 0; v < 8; ++v) st.h[v] = (__bf16)acc[v];
    *(uint4*)(hn + n * HDIM + m0 + half * 8) = st.u;

    __syncthreads();   // single barrier per step (h and x both double-buffered)
  }
}

// ---------------------------------------------------------------------------
// Kernel 2: parallel known pass. grid (B/BT, T/TCHUNK) x 128 threads (4 waves).
//   known_t = A1 @ x_t + A2 @ h_{t-1},  h_{t-1} read from hidden_out (or h0).
// ---------------------------------------------------------------------------
__global__ __launch_bounds__(128, 2)
void rnn_known_kernel(const float* __restrict__ x,
                      const float* __restrict__ h0,
                      const float* __restrict__ A1,  // [K,K]
                      const float* __restrict__ A2,  // [K,H]
                      const float* __restrict__ hidden,  // [T,H,B] (from kernel 1)
                      float* __restrict__ known_out)     // [T,K,B]
{
  __shared__ __align__(16) __bf16 sx[BT * KDIM];
  __shared__ __align__(16) __bf16 sh[BT * HDIM];

  const int tid  = threadIdx.x;
  const int wave = tid >> 5;
  const int lane = tid & 31;
  const int half = lane >> 4;
  const int n    = lane & 15;
  const int b0   = blockIdx.x * BT;
  const int t0   = blockIdx.y * TCHUNK;
  const int m0   = wave * 16;

  bf16x16 aX[2], aH[4];
#pragma unroll
  for (int c = 0; c < 2; ++c) aX[c] = gld_afrag(A1, m0 + n, KDIM, c * 32, half);
#pragma unroll
  for (int c = 0; c < 4; ++c) aH[c] = gld_afrag(A2, m0 + n, HDIM, c * 32, half);

  for (int tt = 0; tt < TCHUNK; ++tt) {
    const int t = t0 + tt;
    const float* hprev = (t == 0) ? (h0 + b0)
                                  : (hidden + (size_t)(t - 1) * HDIM * BDIM + b0);
    const float* xt = x + (size_t)t * KDIM * BDIM + b0;

    // Stage x_t [64x16] and h_{t-1} [128x16] (transpose to [n][k], bf16).
    for (int i = tid; i < (KDIM * BT) / 4; i += 128) {
      int kk = i >> 2, n4 = (i & 3) * 4;
      float4 v = *(const float4*)(xt + (size_t)kk * BDIM + n4);
      sx[(n4 + 0) * KDIM + kk] = (__bf16)v.x;
      sx[(n4 + 1) * KDIM + kk] = (__bf16)v.y;
      sx[(n4 + 2) * KDIM + kk] = (__bf16)v.z;
      sx[(n4 + 3) * KDIM + kk] = (__bf16)v.w;
    }
    for (int i = tid; i < (HDIM * BT) / 4; i += 128) {
      int kk = i >> 2, n4 = (i & 3) * 4;
      float4 v = *(const float4*)(hprev + (size_t)kk * BDIM + n4);
      sh[(n4 + 0) * HDIM + kk] = (__bf16)v.x;
      sh[(n4 + 1) * HDIM + kk] = (__bf16)v.y;
      sh[(n4 + 2) * HDIM + kk] = (__bf16)v.z;
      sh[(n4 + 3) * HDIM + kk] = (__bf16)v.w;
    }
    __syncthreads();

    bf16x16 bx0 = lds_bfrag(sx, n, KDIM, 0, half);
    bf16x16 bx1 = lds_bfrag(sx, n, KDIM, 32, half);
    bf16x16 bh[4];
#pragma unroll
    for (int c = 0; c < 4; ++c) bh[c] = lds_bfrag(sh, n, HDIM, c * 32, half);

    f32x8 accA = {}, accB = {};
    accA = __builtin_amdgcn_wmma_f32_16x16x32_bf16(false, aX[0], false, bx0,
                                                   (short)0, accA, false, false);
    accB = __builtin_amdgcn_wmma_f32_16x16x32_bf16(false, aX[1], false, bx1,
                                                   (short)0, accB, false, false);
    accA = __builtin_amdgcn_wmma_f32_16x16x32_bf16(false, aH[0], false, bh[0],
                                                   (short)0, accA, false, false);
    accB = __builtin_amdgcn_wmma_f32_16x16x32_bf16(false, aH[1], false, bh[1],
                                                   (short)0, accB, false, false);
    accA = __builtin_amdgcn_wmma_f32_16x16x32_bf16(false, aH[2], false, bh[2],
                                                   (short)0, accA, false, false);
    accB = __builtin_amdgcn_wmma_f32_16x16x32_bf16(false, aH[3], false, bh[3],
                                                   (short)0, accB, false, false);
    f32x8 acc = accA + accB;

    float* outp = known_out + (size_t)t * KDIM * BDIM + (size_t)m0 * BDIM + b0;
#pragma unroll
    for (int v = 0; v < 8; ++v)
      outp[(size_t)(half * 8 + v) * BDIM + n] = acc[v];

    __syncthreads();  // staging buffers reused next tt
  }
}

extern "C" void kernel_launch(void* const* d_in, const int* in_sizes, int n_in,
                              void* d_out, int out_size, void* d_ws, size_t ws_size,
                              hipStream_t stream) {
  const float* x  = (const float*)d_in[0];
  const float* h0 = (const float*)d_in[1];
  const float* A1 = (const float*)d_in[2];
  const float* A2 = (const float*)d_in[3];
  const float* A3 = (const float*)d_in[4];
  const float* A4 = (const float*)d_in[5];

  float* known_out  = (float*)d_out;                                   // [T,K,B]
  float* hidden_out = (float*)d_out + (size_t)T_STEPS * KDIM * BDIM;   // [T,H,B]

  // Phase 1: sequential hidden scan (latency-critical, 8 WGPs).
  rnn_hidden_scan<<<dim3(BDIM / BT), dim3(256), 0, stream>>>(
      x, h0, A3, A4, hidden_out);

  // Phase 2: embarrassingly parallel known pass (fills the chip).
  rnn_known_kernel<<<dim3(BDIM / BT, T_STEPS / TCHUNK), dim3(128), 0, stream>>>(
      x, h0, A1, A2, hidden_out, known_out);
}
